// SRModel_12343736009411
// MI455X (gfx1250) — matrix operand order repaired and checked by reading, not compiled, not run on hardware
//
#include <hip/hip_runtime.h>
#include <math.h>
#include <stdint.h>

// CDNA5 WMMA operand types (wave32): 16 bf16 per lane (8 VGPRs), 8 f32 accum VGPRs.
typedef __attribute__((ext_vector_type(16))) __bf16 v16bf;
typedef __attribute__((ext_vector_type(8)))  float  v8f;

namespace {

constexpr int kB   = 16;         // batch
constexpr int kF   = 128;        // freq bins
constexpr int kT0  = 1024;       // input time
constexpr int kC   = 32;         // cnn channels
constexpr int kH   = 64;         // F/2
constexpr int kT   = 512;        // T0/2
constexpr int kDin = kC * kH;    // 2048
constexpr int kHid = 512;
constexpr int kD   = 2 * kHid;   // 1024
constexpr int kM   = kT * kB;    // 8192 sequence rows
constexpr int kV   = 29;         // vocab + 1

// ---------------- front-end conv (stride-2, 1->32) ----------------
__global__ void conv_stem_kernel(const float* __restrict__ x,
                                 const float* __restrict__ w,
                                 const float* __restrict__ bias,
                                 float* __restrict__ out)
{
    int tid = blockIdx.x * blockDim.x + threadIdx.x;
    const int total = kB * kC * kH * kT;
    if (tid >= total) return;
    int j = tid % kT;  int t2 = tid / kT;
    int i = t2 % kH;   t2 /= kH;
    int o = t2 % kC;   int b = t2 / kC;
    float s = bias[o];
    #pragma unroll
    for (int di = 0; di < 3; ++di) {
        int ii = 2 * i - 1 + di;
        if (ii < 0 || ii >= kF) continue;
        #pragma unroll
        for (int dj = 0; dj < 3; ++dj) {
            int jj = 2 * j - 1 + dj;
            if (jj < 0 || jj >= kT0) continue;
            s += x[((size_t)b * kF + ii) * kT0 + jj] * w[o * 9 + di * 3 + dj];
        }
    }
    out[tid] = s;
}

// ---------------- residual 3x3 conv + BN (+residual) (+relu) ----------------
__global__ void conv_res_kernel(const float* __restrict__ in,
                                const float* __restrict__ w,
                                const float* __restrict__ bias,
                                const float* __restrict__ bng,
                                const float* __restrict__ bnb,
                                const float* __restrict__ bnm,
                                const float* __restrict__ bnv,
                                const float* __restrict__ resid,  // may be null
                                float* __restrict__ out,
                                int relu)
{
    int tid = blockIdx.x * blockDim.x + threadIdx.x;
    const int total = kB * kC * kH * kT;
    if (tid >= total) return;
    int j = tid % kT;  int t2 = tid / kT;
    int i = t2 % kH;   t2 /= kH;
    int o = t2 % kC;   int b = t2 / kC;
    float s = bias[o];
    for (int c = 0; c < kC; ++c) {
        const float* ip = in + ((size_t)b * kC + c) * kH * kT;
        const float* wp = w + ((size_t)o * kC + c) * 9;
        #pragma unroll
        for (int di = 0; di < 3; ++di) {
            int ii = i - 1 + di;
            if (ii < 0 || ii >= kH) continue;
            #pragma unroll
            for (int dj = 0; dj < 3; ++dj) {
                int jj = j - 1 + dj;
                if (jj < 0 || jj >= kT) continue;
                s += ip[(size_t)ii * kT + jj] * wp[di * 3 + dj];
            }
        }
    }
    float scale = bng[o] * rsqrtf(bnv[o] + 1e-5f);
    float v = s * scale + (bnb[o] - bnm[o] * scale);
    if (resid) v += resid[tid];
    if (relu) v = fmaxf(v, 0.f);
    out[tid] = v;
}

// ---------------- (B,C,H,T) -> (T*B, C*H) sequence matrix ----------------
__global__ void to_seq_kernel(const float* __restrict__ conv, float* __restrict__ X)
{
    int tid = blockIdx.x * blockDim.x + threadIdx.x;
    const int total = kB * kC * kH * kT;
    if (tid >= total) return;
    int t = tid % kT;  int t2 = tid / kT;
    int hh = t2 % kH;  t2 /= kH;
    int c = t2 % kC;   int b = t2 / kC;
    X[((size_t)t * kB + b) * kDin + (c * kH + hh)] = conv[tid];
}

// ---------------- layer norm over last dim; optional (t,b)->(b,t) row remap ----------------
__global__ void layernorm_kernel(const float* __restrict__ X,
                                 const float* __restrict__ g,
                                 const float* __restrict__ beta,
                                 float* __restrict__ Y,
                                 int D, int transposeTB)
{
    int row = blockIdx.x;
    const float* xr = X + (size_t)row * D;
    float s = 0.f, s2 = 0.f;
    for (int i = threadIdx.x; i < D; i += blockDim.x) {
        float v = xr[i]; s += v; s2 += v * v;
    }
    #pragma unroll
    for (int off = 16; off > 0; off >>= 1) {
        s  += __shfl_xor(s,  off, 32);
        s2 += __shfl_xor(s2, off, 32);
    }
    __shared__ float ws[8], ws2[8];
    __shared__ float smu, srs;
    int lane = threadIdx.x & 31, wv = threadIdx.x >> 5;
    if (lane == 0) { ws[wv] = s; ws2[wv] = s2; }
    __syncthreads();
    if (threadIdx.x == 0) {
        float a = 0.f, a2 = 0.f;
        int nw = blockDim.x >> 5;
        for (int q = 0; q < nw; ++q) { a += ws[q]; a2 += ws2[q]; }
        float mu = a / (float)D;
        float var = a2 / (float)D - mu * mu;
        smu = mu; srs = rsqrtf(var + 1e-5f);
    }
    __syncthreads();
    float mu = smu, rs = srs;
    size_t rowOut = row;
    if (transposeTB) {
        int t = row / kB, b = row % kB;
        rowOut = (size_t)b * kT + t;
    }
    for (int i = threadIdx.x; i < D; i += blockDim.x)
        Y[rowOut * D + i] = (xr[i] - mu) * rs * g[i] + beta[i];
}

// ---------------- pack A (MxK f32, row-major) into WMMA A-fragment layout ----------------
// A 16x32 bf16 fragment, lane L: m = L%16, h = L/16; slot e (0..15):
//   k = 8*h + e            (e < 8)
//   k = 16 + 8*h + (e-8)   (e >= 8)
__global__ void pack_a_kernel(const float* __restrict__ X, __bf16* __restrict__ pA,
                              int M, int K)
{
    int tid = blockIdx.x * blockDim.x + threadIdx.x;
    int KT = K >> 5;
    int total = (M >> 4) * KT * 32;
    if (tid >= total) return;
    int lane = tid & 31;
    int frag = tid >> 5;           // mt*KT + kt
    int kt = frag % KT;
    int mt = frag / KT;
    int m = (mt << 4) + (lane & 15);
    int h = lane >> 4;
    const float* xr = X + (size_t)m * K + (kt << 5);
    v16bf outv;
    #pragma unroll
    for (int e = 0; e < 16; ++e) {
        int k = 8 * h + e + ((e >= 8) ? 8 : 0);
        outv[e] = (__bf16)xr[k];
    }
    *((v16bf*)(pA) + tid) = outv;
}

// ---------------- pack B (KxN f32, row-major) into group-major WMMA B fragments ----------
// B 32x16 bf16 fragment, lane L: n = L%16, h = L/16; slot e: k = 8*h + e (e<8), 16 + 8*h + (e-8) (e>=8). Zero-pad n>=N.
// Output order: [ntGroup][kt][jj in group][lane] so a GEMM block's JT fragments for one
// K-step are one contiguous JT*1024-byte chunk (async-DMA friendly).
__global__ void pack_b_kernel(const float* __restrict__ W, __bf16* __restrict__ pB,
                              int K, int N, int Npad, int JT)
{
    int tid = blockIdx.x * blockDim.x + threadIdx.x;
    int KT = K >> 5;
    int total = (Npad >> 4) * KT * 32;
    if (tid >= total) return;
    int lane = tid & 31;
    int frag = tid >> 5;           // nt*KT + kt
    int kt = frag % KT;
    int nt = frag / KT;
    int n = (nt << 4) + (lane & 15);
    int h = lane >> 4;
    v16bf outv;
    #pragma unroll
    for (int e = 0; e < 16; ++e) {
        int k = (kt << 5) + 8 * h + e + ((e >= 8) ? 8 : 0);
        outv[e] = (n < N) ? (__bf16)W[(size_t)k * N + n] : (__bf16)0.f;
    }
    int grp = nt / JT, jj = nt % JT;
    size_t chunk = ((size_t)grp * KT + kt) * JT + jj;     // 512-elem fragments
    *((v16bf*)(pB) + chunk * 32 + lane) = outv;
}

// ---------------- bf16 WMMA GEMM with async-to-LDS B staging ----------------
// Block = 8 waves; wave w owns 32 rows (2 A fragments), all waves share blockIdx.x's JT
// B fragments per K-step. B is staged through a 4-deep LDS ring with
// GLOBAL_LOAD_ASYNC_TO_LDS (ASYNCcnt) issued one step ahead; one barrier per step.
// A fragments are register double-buffered from global. Steady-state K-loop is branch-free
// (last step peeled) and pinned to no-unroll to stay within VGPR budget (no spills).
// Requires M % 256 == 0 and grid to exactly cover M/32 x Npad/(16*JT) tiles.
template <int JT>
__global__ void gemm_bf16_wmma(const __bf16* __restrict__ pA,
                               const __bf16* __restrict__ pB,
                               float* __restrict__ C,
                               int M, int N, int K, int ldc, int relu)
{
    constexpr int kChunk = JT * 1024;            // bytes of B per K-step
    __shared__ __align__(16) char smem[4 * kChunk];

    const int lane = threadIdx.x & 31;
    const int wave = threadIdx.x >> 5;
    const int tid  = threadIdx.x;
    const int mt2  = blockIdx.y * (blockDim.x >> 5) + wave;   // 32-row tile index
    const int KT   = K >> 5;
    const int nt0  = blockIdx.x * JT;

    const v16bf* A0 = (const v16bf*)pA + ((size_t)(2 * mt2) * KT) * 32 + lane;
    const v16bf* A1 = A0 + (size_t)KT * 32;

    // async B copy: 256 threads move kChunk bytes (16B each for JT=4, 8B for JT=2)
    constexpr int kPerThread = kChunk / 256;
    const char* bsrc = (const char*)pB + (size_t)blockIdx.x * KT * kChunk + (size_t)tid * kPerThread;
    const uint32_t ldsBase = (uint32_t)(uintptr_t)(&smem[0]) + (uint32_t)(tid * kPerThread);

    auto issue_copy = [&](int kt) {
        const char* g = bsrc + (size_t)kt * kChunk;
        uint32_t l = ldsBase + (uint32_t)((kt & 3) * kChunk);
        if (JT == 4)
            asm volatile("global_load_async_to_lds_b128 %0, %1, off" :: "v"(l), "v"(g) : "memory");
        else
            asm volatile("global_load_async_to_lds_b64 %0, %1, off" :: "v"(l), "v"(g) : "memory");
    };

    const v8f z = {0.f, 0.f, 0.f, 0.f, 0.f, 0.f, 0.f, 0.f};
    v8f acc0[JT], acc1[JT];
    #pragma unroll
    for (int j = 0; j < JT; ++j) { acc0[j] = z; acc1[j] = z; }

    issue_copy(0);
    v16bf a0c = A0[0];
    v16bf a1c = A1[0];

    // steady state: branch-free body; copy(kt+1) in flight while computing kt
    #pragma unroll 1
    for (int kt = 0; kt < KT - 1; ++kt) {
        issue_copy(kt + 1);
        asm volatile("s_wait_asynccnt 0x1" ::: "memory");   // copy(kt) done
        __syncthreads();                                    // all waves' slices visible

        const v16bf* Bl = (const v16bf*)(smem + (size_t)(kt & 3) * kChunk);
        v16bf bfr[JT];
        #pragma unroll
        for (int j = 0; j < JT; ++j) bfr[j] = Bl[j * 32 + lane];

        // prefetch next A fragments while the WMMAs run
        v16bf a0n = A0[(size_t)(kt + 1) * 32];
        v16bf a1n = A1[(size_t)(kt + 1) * 32];

        #pragma unroll
        for (int j = 0; j < JT; ++j) {
            acc0[j] = __builtin_amdgcn_wmma_f32_16x16x32_bf16(false, a0c, false, bfr[j], (short)0, acc0[j], false, false);
            acc1[j] = __builtin_amdgcn_wmma_f32_16x16x32_bf16(false, a1c, false, bfr[j], (short)0, acc1[j], false, false);
        }
        a0c = a0n; a1c = a1n;
    }

    // epilogue: kt = KT-1
    {
        const int kt = KT - 1;
        asm volatile("s_wait_asynccnt 0x0" ::: "memory");
        __syncthreads();
        const v16bf* Bl = (const v16bf*)(smem + (size_t)(kt & 3) * kChunk);
        v16bf bfr[JT];
        #pragma unroll
        for (int j = 0; j < JT; ++j) bfr[j] = Bl[j * 32 + lane];
        #pragma unroll
        for (int j = 0; j < JT; ++j) {
            acc0[j] = __builtin_amdgcn_wmma_f32_16x16x32_bf16(false, a0c, false, bfr[j], (short)0, acc0[j], false, false);
            acc1[j] = __builtin_amdgcn_wmma_f32_16x16x32_bf16(false, a1c, false, bfr[j], (short)0, acc1[j], false, false);
        }
    }

    // C/D layout: lane L: n = L%16, h = L/16; VGPR r holds row M = r + 8*h.
    const int n = lane & 15, h = lane >> 4;
    #pragma unroll
    for (int j = 0; j < JT; ++j) {
        const int col = ((nt0 + j) << 4) + n;
        if (col >= N) continue;
        #pragma unroll
        for (int r = 0; r < 8; ++r) {
            int row0 = ((2 * mt2) << 4) + r + (h << 3);
            float v0 = acc0[j][r];
            float v1 = acc1[j][r];
            if (relu) { v0 = fmaxf(v0, 0.f); v1 = fmaxf(v1, 0.f); }
            C[(size_t)row0 * ldc + col] = v0;
            C[(size_t)(row0 + 16) * ldc + col] = v1;
        }
    }
}

// ---------------- bidirectional SRU recurrence (tanh, highway) ----------------
// U layout: (dir, T*B rows = t*B+b, 4*hid cols = [xt|fp|rp|xp]). Dir 1 scans t backwards
// (equivalent to reversing the input), writing h directly at position t.
__global__ void sru_scan_kernel(const float* __restrict__ U,
                                const float* __restrict__ vprm,   // (2,2,hid)
                                const float* __restrict__ bprm,   // (2,2,hid)
                                float* __restrict__ ynext)        // (T*B, 2*hid)
{
    int tid = blockIdx.x * blockDim.x + threadIdx.x;
    const int total = 2 * kB * kHid;
    if (tid >= total) return;
    int dir = tid / (kB * kHid);
    int rem = tid % (kB * kHid);
    int b = rem / kHid;
    int j = rem % kHid;
    const float* Ud = U + (size_t)dir * kM * 4 * kHid;
    float vf = vprm[(dir * 2 + 0) * kHid + j];
    float vr = vprm[(dir * 2 + 1) * kHid + j];
    float bf = bprm[(dir * 2 + 0) * kHid + j];
    float br = bprm[(dir * 2 + 1) * kHid + j];
    float c = 0.f;
    for (int s = 0; s < kT; ++s) {
        int t = dir ? (kT - 1 - s) : s;
        size_t ro = ((size_t)t * kB + b) * (4 * (size_t)kHid);
        float xt = Ud[ro + j];
        float fp = Ud[ro + kHid + j];
        float rp = Ud[ro + 2 * kHid + j];
        float xp = Ud[ro + 3 * kHid + j];
        float f = 1.f / (1.f + expf(-(fp + vf * c + bf)));
        float r = 1.f / (1.f + expf(-(rp + vr * c + br)));
        c = f * c + (1.f - f) * xt;
        float hv = r * tanhf(c) + (1.f - r) * xp;
        ynext[((size_t)t * kB + b) * (2 * (size_t)kHid) + dir * kHid + j] = hv;
    }
}

} // namespace

extern "C" void kernel_launch(void* const* d_in, const int* in_sizes, int n_in,
                              void* d_out, int out_size, void* d_ws, size_t ws_size,
                              hipStream_t stream)
{
    (void)in_sizes; (void)n_in; (void)out_size; (void)ws_size;
    const float* x       = (const float*)d_in[0];
    const float* cnn_w   = (const float*)d_in[1];
    const float* cnn_b   = (const float*)d_in[2];
    const float* c1w     = (const float*)d_in[3];
    const float* c1b     = (const float*)d_in[4];
    const float* bn1g    = (const float*)d_in[5];
    const float* bn1b    = (const float*)d_in[6];
    const float* bn1m    = (const float*)d_in[7];
    const float* bn1v    = (const float*)d_in[8];
    const float* c2w     = (const float*)d_in[9];
    const float* c2b     = (const float*)d_in[10];
    const float* bn2g    = (const float*)d_in[11];
    const float* bn2b    = (const float*)d_in[12];
    const float* bn2m    = (const float*)d_in[13];
    const float* bn2v    = (const float*)d_in[14];
    const float* proj_W  = (const float*)d_in[15];
    const float* sru_W   = (const float*)d_in[16];
    const float* sru_v   = (const float*)d_in[17];
    const float* sru_b   = (const float*)d_in[18];
    const float* sru_lng = (const float*)d_in[19];
    const float* sru_lnb = (const float*)d_in[20];
    const float* cls_lng = (const float*)d_in[21];
    const float* cls_lnb = (const float*)d_in[22];
    const float* cls_W1  = (const float*)d_in[23];
    const float* cls_W2  = (const float*)d_in[24];
    float* out = (float*)d_out;

    // ---- workspace carve-up (U reuses the conv ping/pong region, free by then) ----
    char* ws = (char*)d_ws;
    const size_t convBytes = (size_t)kB * kC * kH * kT * 4;   // 64 MiB
    float*  cb0  = (float*)(ws + 0 * convBytes);
    float*  cb1  = (float*)(ws + 1 * convBytes);
    float*  cb2  = (float*)(ws + 2 * convBytes);
    float*  Xmat = (float*)(ws + 3 * convBytes);                        // up to 8192x2048 f32
    __bf16* pA   = (__bf16*)((char*)Xmat + (size_t)kM * kDin * 4);      // 8192x2048 bf16
    __bf16* pB   = (__bf16*)((char*)pA + (size_t)kM * kDin * 2);        // up to 2048x2048 bf16
    float*  y0   = (float*)((char*)pB + (size_t)2048 * 2048 * 2);       // 8192x1024 f32
    float*  y1   = (float*)((char*)y0 + (size_t)kM * kD * 4);
    float*  clsH = (float*)((char*)y1 + (size_t)kM * kD * 4);           // 8192x512 f32
    float*  Ubuf = (float*)(ws + 0);                                    // 2 x 8192 x 2048 f32

    const int totalConv = kB * kC * kH * kT;
    const int thr = 256;
    const int convBlocks = (totalConv + thr - 1) / thr;

    auto packA = [&](const float* Xp, int M, int K) {
        int total = (M / 16) * (K / 32) * 32;
        pack_a_kernel<<<(total + thr - 1) / thr, thr, 0, stream>>>(Xp, pA, M, K);
    };
    auto packB = [&](const float* Wp, int K, int N, int Npad, int JT) {
        int total = (Npad / 16) * (K / 32) * 32;
        pack_b_kernel<<<(total + thr - 1) / thr, thr, 0, stream>>>(Wp, pB, K, N, Npad, JT);
    };
    // JT=4 path: Npad multiple of 64; JT=2 path for the 32-col classifier head.
    auto gemm4 = [&](float* Cp, int M, int N, int Npad, int K, int ldc, int relu) {
        dim3 grid(Npad / 64, M / 256);
        gemm_bf16_wmma<4><<<grid, 256, 0, stream>>>(pA, pB, Cp, M, N, K, ldc, relu);
    };
    auto gemm2 = [&](float* Cp, int M, int N, int Npad, int K, int ldc, int relu) {
        dim3 grid(Npad / 32, M / 256);
        gemm_bf16_wmma<2><<<grid, 256, 0, stream>>>(pA, pB, Cp, M, N, K, ldc, relu);
    };

    // 1) front-end conv
    conv_stem_kernel<<<convBlocks, thr, 0, stream>>>(x, cnn_w, cnn_b, cb0);

    // 2) residual CNN blocks
    float* cur = cb0; float* f1 = cb1; float* f2 = cb2;
    for (int i = 0; i < 3; ++i) {
        size_t wOff = (size_t)i * kC * kC * 9;
        conv_res_kernel<<<convBlocks, thr, 0, stream>>>(
            cur, c1w + wOff, c1b + i * kC, bn1g + i * kC, bn1b + i * kC,
            bn1m + i * kC, bn1v + i * kC, nullptr, f1, 1);
        conv_res_kernel<<<convBlocks, thr, 0, stream>>>(
            f1, c2w + wOff, c2b + i * kC, bn2g + i * kC, bn2b + i * kC,
            bn2m + i * kC, bn2v + i * kC, cur, f2, 1);
        float* ncur = f2; f2 = f1; f1 = cur; cur = ncur;
    }

    // 3) reshape to (T*B, C*H) and project to D=1024 (WMMA GEMM)
    to_seq_kernel<<<convBlocks, thr, 0, stream>>>(cur, Xmat);
    packA(Xmat, kM, kDin);
    packB(proj_W, kDin, kD, kD, 4);
    gemm4(y0, kM, kD, kD, kDin, kD, 0);

    // 4) SRU layers: LN -> U = xn @ W (both dirs, WMMA) -> sequential scan
    float* ycur = y0; float* ynext = y1;
    for (int l = 0; l < 4; ++l) {
        layernorm_kernel<<<kM, 256, 0, stream>>>(ycur, sru_lng + (size_t)l * kD,
                                                 sru_lnb + (size_t)l * kD, Xmat, kD, 0);
        packA(Xmat, kM, kD);
        for (int dir = 0; dir < 2; ++dir) {
            packB(sru_W + ((size_t)(l * 2 + dir)) * kD * (4 * kHid), kD, 4 * kHid, 4 * kHid, 4);
            gemm4(Ubuf + (size_t)dir * kM * 4 * kHid, kM, 4 * kHid, 4 * kHid, kD, 4 * kHid, 0);
        }
        sru_scan_kernel<<<(2 * kB * kHid + thr - 1) / thr, thr, 0, stream>>>(
            Ubuf, sru_v + (size_t)l * 2 * 2 * kHid, sru_b + (size_t)l * 2 * 2 * kHid, ynext);
        float* tmp = ycur; ycur = ynext; ynext = tmp;
    }

    // 5) classifier: LN (with (t,b)->(b,t) remap) -> relu(X@W1) -> X@W2
    layernorm_kernel<<<kM, 256, 0, stream>>>(ycur, cls_lng, cls_lnb, Xmat, kD, 1);
    packA(Xmat, kM, kD);
    packB(cls_W1, kD, kHid, kHid, 4);
    gemm4(clsH, kM, kHid, kHid, kD, kHid, 1);

    packA(clsH, kM, kHid);
    packB(cls_W2, kHid, kV, 32, 2);    // pad N=29 -> 32 with zeros, guard on store
    gemm2(out, kM, kV, 32, kHid, kV, 0);
}
